// QuantQwen3DecoderLayer_89111981457672
// MI455X (gfx1250) — compile-verified, matrix operand order checked by reading, not run on hardware
//
#include <hip/hip_runtime.h>
#include <hip/hip_bf16.h>

// ---------------- problem constants ----------------
constexpr int Bc  = 2;
constexpr int Sc  = 2048;
constexpr int Dc  = 4096;
constexpr int Hc  = 32;
constexpr int KVc = 8;
constexpr int HDc = 128;
constexpr int Ic  = 12288;
constexpr int Mc  = Bc * Sc;          // 4096 token rows
constexpr int QNc = Hc * HDc;         // 4096
constexpr int KNc = KVc * HDc;        // 1024
constexpr float RSQRT_HD = 0.08838834764831845f;  // 1/sqrt(128)
constexpr float THETA = 1000000.0f;

typedef __attribute__((ext_vector_type(16))) _Float16 v16h;
typedef __attribute__((ext_vector_type(8)))  float    v8f;

__device__ __forceinline__ v8f wmma16(v16h a, v16h b, v8f c) {
    return __builtin_amdgcn_wmma_f32_16x16x32_f16(
        /*neg_a=*/false, a, /*neg_b=*/false, b,
        /*c_mod=*/(short)0, c, /*reuse_a=*/false, /*reuse_b=*/false);
}

// ---- gfx1250 async global->LDS copy (16 B per lane), tracked by ASYNCcnt ----
__device__ __forceinline__ void async_cp16(const _Float16* g, _Float16* l) {
    unsigned loff = (unsigned)(size_t)(void*)l;   // LDS aperture: low 32 bits = LDS byte offset
    asm volatile("global_load_async_to_lds_b128 %0, %1, off"
                 :: "v"(loff), "v"(g) : "memory");
}
__device__ __forceinline__ void wait_async6() {
    asm volatile("s_wait_asynccnt 0x6" ::: "memory");
}
__device__ __forceinline__ void wait_async0() {
    asm volatile("s_wait_asynccnt 0x0" ::: "memory");
}

// ---------------- block reductions ----------------
__device__ __forceinline__ float blockSum(float v, float* sbuf) {
    int t = threadIdx.x;
    sbuf[t] = v; __syncthreads();
    for (int o = blockDim.x >> 1; o > 0; o >>= 1) {
        if (t < o) sbuf[t] += sbuf[t + o];
        __syncthreads();
    }
    float r = sbuf[0]; __syncthreads();
    return r;
}

__device__ __forceinline__ void blockMinMax(float& mn, float& mx, float* smn, float* smx) {
    int t = threadIdx.x;
    smn[t] = mn; smx[t] = mx; __syncthreads();
    for (int o = blockDim.x >> 1; o > 0; o >>= 1) {
        if (t < o) {
            smn[t] = fminf(smn[t], smn[t + o]);
            smx[t] = fmaxf(smx[t], smx[t + o]);
        }
        __syncthreads();
    }
    mn = smn[0]; mx = smx[0]; __syncthreads();
}

// asymmetric 4-bit fake-quant params (matches reference fake_quant)
__device__ __forceinline__ void qparams(float mn, float mx, float& scale, float& zp, float& inv) {
    mn = fminf(mn, 0.0f);
    mx = fmaxf(mx, 0.0f);
    scale = fmaxf((mx - mn) * (1.0f / 15.0f), 1e-8f);
    zp    = roundf(-mn / scale);
    inv   = 1.0f / scale;
}
__device__ __forceinline__ float qdq(float x, float scale, float zp, float inv) {
    float q = fminf(fmaxf(roundf(x * inv) + zp, 0.0f), 15.0f);
    return (q - zp) * scale;
}

// ---------------- weight quantization: per-output-channel, fp32 -> quantdequant f16 -----
// one block per weight row; row length K
__global__ void k_quant_weight(const float* __restrict__ w, _Float16* __restrict__ wq, int K) {
    __shared__ float smn[256], smx[256];
    const size_t row = blockIdx.x;
    const float* src = w + row * (size_t)K;
    _Float16*    dst = wq + row * (size_t)K;
    float mn = 3e38f, mx = -3e38f;
    for (int i = threadIdx.x; i < K; i += blockDim.x) {
        float v = src[i];
        mn = fminf(mn, v); mx = fmaxf(mx, v);
    }
    blockMinMax(mn, mx, smn, smx);
    float scale, zp, inv;
    qparams(mn, mx, scale, zp, inv);
    for (int i = threadIdx.x; i < K; i += blockDim.x)
        dst[i] = (_Float16)qdq(src[i], scale, zp, inv);
}

// ---------------- RMSNorm + per-token fake-quant -> f16 ----------------
// one block (256 thr) per token row of length D
__global__ void k_rmsnorm_quant(const float* __restrict__ x, const float* __restrict__ w,
                                _Float16* __restrict__ out, int D) {
    __shared__ float sbuf[256];
    __shared__ float smn[256], smx[256];
    const size_t row = blockIdx.x;
    const float* src = x + row * (size_t)D;
    _Float16*    dst = out + row * (size_t)D;
    float ss = 0.0f;
    for (int i = threadIdx.x; i < D; i += blockDim.x) {
        float v = src[i];
        ss += v * v;
    }
    ss = blockSum(ss, sbuf);
    const float rinv = rsqrtf(ss / (float)D + 1e-6f);
    float mn = 3e38f, mx = -3e38f;
    for (int i = threadIdx.x; i < D; i += blockDim.x) {
        float v = src[i] * rinv * w[i];
        mn = fminf(mn, v); mx = fmaxf(mx, v);
    }
    blockMinMax(mn, mx, smn, smx);
    float scale, zp, inv;
    qparams(mn, mx, scale, zp, inv);
    for (int i = threadIdx.x; i < D; i += blockDim.x) {
        float v = src[i] * rinv * w[i];
        dst[i] = (_Float16)qdq(v, scale, zp, inv);
    }
}

// ---------------- WMMA TN GEMM with async-LDS staging ----------------
// C[M,N] f32 = A[M,K] f16 * W[N,K] f16
// block: 128 threads (4 waves), block tile 64(M) x 128(N), k-slab 32.
// k-slabs (A: 64x32, W: 128x32) are DMA'd into LDS with
// global_load_async_to_lds_b128 (ASYNCcnt), double-buffered so the next slab's
// copies overlap the current 8-WMMA pack. Each wave computes a 32x64 tile from
// ds_load_b128 fragments: 2 A-frags x 4 B-frags -> 8 WMMAs per slab.
// grid: (N/128, M/64)
__global__ void k_gemm_tn(const _Float16* __restrict__ A, const _Float16* __restrict__ W,
                          float* __restrict__ C, int N, int K) {
    __shared__ __align__(32) _Float16 sA[2][64 * 32];    // 8 KB
    __shared__ __align__(32) _Float16 sW[2][128 * 32];   // 16 KB
    const int tid  = threadIdx.x;
    const int lane = tid & 31;
    const int wave = tid >> 5;
    const int half = lane >> 4;
    const int l16  = lane & 15;
    const int m0blk = blockIdx.y * 64;
    const int n0blk = blockIdx.x * 128;
    const int mlocal = (wave & 1) * 32;   // wave's M offset within block tile
    const int nlocal = (wave >> 1) * 64;  // wave's N offset within block tile

    // per-thread copy assignments (16 B chunks): A = 256 chunks, W = 512 chunks
    const int arow0 = (tid + 0 * 128) >> 2, asub0 = (tid + 0 * 128) & 3;
    const int arow1 = (tid + 1 * 128) >> 2, asub1 = (tid + 1 * 128) & 3;

    auto stage = [&](int buf, int k) {
        async_cp16(A + (size_t)(m0blk + arow0) * K + k + asub0 * 8, &sA[buf][arow0 * 32 + asub0 * 8]);
        async_cp16(A + (size_t)(m0blk + arow1) * K + k + asub1 * 8, &sA[buf][arow1 * 32 + asub1 * 8]);
#pragma unroll
        for (int i = 0; i < 4; ++i) {
            const int c = tid + i * 128;
            const int row = c >> 2, sub = c & 3;
            async_cp16(W + (size_t)(n0blk + row) * K + k + sub * 8, &sW[buf][row * 32 + sub * 8]);
        }
    };

    v8f acc[2][4];
#pragma unroll
    for (int i = 0; i < 2; ++i)
#pragma unroll
        for (int j = 0; j < 4; ++j) acc[i][j] = (v8f){};

    stage(0, 0);
    int buf = 0;
    for (int k = 0; k < K; k += 32) {
        const bool more = (k + 32) < K;
        if (more) {
            stage(buf ^ 1, k + 32);
            __builtin_prefetch(A + (size_t)(m0blk + l16) * K + k + 256, 0, 1);  // global_prefetch_b8
            __builtin_prefetch(W + (size_t)(n0blk + l16) * K + k + 256, 0, 1);
            wait_async6();          // previous slab landed; next slab (6 copies) still in flight
        } else {
            wait_async0();
        }
        __syncthreads();

        const _Float16* sa = &sA[buf][(mlocal + l16) * 32 + half * 16];
        const _Float16* sw = &sW[buf][(nlocal + l16) * 32 + half * 16];
        v16h a0 = *(const v16h*)(sa);
        v16h a1 = *(const v16h*)(sa + 16 * 32);
        v16h b0 = *(const v16h*)(sw);
        v16h b1 = *(const v16h*)(sw + 16 * 32);
        v16h b2 = *(const v16h*)(sw + 32 * 32);
        v16h b3 = *(const v16h*)(sw + 48 * 32);

        acc[0][0] = wmma16(a0, b0, acc[0][0]);
        acc[0][1] = wmma16(a0, b1, acc[0][1]);
        acc[0][2] = wmma16(a0, b2, acc[0][2]);
        acc[0][3] = wmma16(a0, b3, acc[0][3]);
        acc[1][0] = wmma16(a1, b0, acc[1][0]);
        acc[1][1] = wmma16(a1, b1, acc[1][1]);
        acc[1][2] = wmma16(a1, b2, acc[1][2]);
        acc[1][3] = wmma16(a1, b3, acc[1][3]);

        __syncthreads();   // all waves done reading buf before it is re-targeted
        buf ^= 1;
    }

#pragma unroll
    for (int i = 0; i < 2; ++i) {
        float* cp = C + (size_t)(m0blk + mlocal + i * 16 + half * 8) * N + n0blk + nlocal + l16;
#pragma unroll
        for (int j = 0; j < 4; ++j)
#pragma unroll
            for (int r = 0; r < 8; ++r)
                cp[(size_t)r * N + j * 16] = acc[i][j][r];
    }
}

// ---------------- RoPE + per-row fake-quant; writes Q [B,H,S,HD] / K [B,KV,S,HD] f16 ----
// grid: (S, H+KV, B), block 128 (= HD threads)
__global__ void k_rope_quant(const float* __restrict__ qf, const float* __restrict__ kf,
                             _Float16* __restrict__ Qo, _Float16* __restrict__ Ko) {
    __shared__ float smn[128], smx[128];
    const int d = threadIdx.x;
    const int s = blockIdx.x;
    const int hh = blockIdx.y;
    const int b = blockIdx.z;
    const float* src;
    _Float16* dst;
    if (hh < Hc) {
        src = qf + ((size_t)(b * Sc + s)) * QNc + hh * HDc;
        dst = Qo + (((size_t)b * Hc + hh) * Sc + s) * HDc;
    } else {
        const int kv = hh - Hc;
        src = kf + ((size_t)(b * Sc + s)) * KNc + kv * HDc;
        dst = Ko + (((size_t)b * KVc + kv) * Sc + s) * HDc;
    }
    const float v = src[d];
    const int i = d & 63;
    const float invf = __powf(THETA, -(2.0f * (float)i) / 128.0f);
    const float ang = (float)s * invf;
    const float c = __cosf(ang), sn = __sinf(ang);
    const float rot = (d < 64) ? -src[d + 64] : src[d - 64];
    const float r = v * c + rot * sn;
    float mn = r, mx = r;
    blockMinMax(mn, mx, smn, smx);
    float scale, zp, inv;
    qparams(mn, mx, scale, zp, inv);
    dst[d] = (_Float16)qdq(r, scale, zp, inv);
}

// ---------------- V: per-row fake-quant + transpose -> Vt [B,KV,HD,S] f16 ----------------
// grid: (S, KV, B), block 128
__global__ void k_vq_transpose(const float* __restrict__ vf, _Float16* __restrict__ Vt) {
    __shared__ float smn[128], smx[128];
    const int d = threadIdx.x;
    const int s = blockIdx.x;
    const int kv = blockIdx.y;
    const int b = blockIdx.z;
    const float v = vf[(size_t)(b * Sc + s) * KNc + kv * HDc + d];
    float mn = v, mx = v;
    blockMinMax(mn, mx, smn, smx);
    float scale, zp, inv;
    qparams(mn, mx, scale, zp, inv);
    Vt[(((size_t)b * KVc + kv) * HDc + d) * Sc + s] = (_Float16)qdq(v, scale, zp, inv);
}

// ---------------- flash attention with WMMA (QK^T and PV), causal ----------------
// grid: (S/64, H, B), block 128 (4 waves, each wave = one 16-row q tile)
__global__ void k_attn(const _Float16* __restrict__ Q, const _Float16* __restrict__ Km,
                       const _Float16* __restrict__ Vt, _Float16* __restrict__ O) {
    __shared__ __align__(32) _Float16 lp[4][16 * 32];  // per-wave P staging tile
    const int lane = threadIdx.x & 31;
    const int wave = threadIdx.x >> 5;
    const int half = lane >> 4;
    const int l16  = lane & 15;
    const int qt = blockIdx.x * 4 + wave;
    const int h  = blockIdx.y;
    const int b  = blockIdx.z;
    const int kvh = h / (Hc / KVc);
    _Float16* lpw = lp[wave];

    // preload Q fragments (16 rows x 128 hd) once
    const _Float16* qbase = Q + (((size_t)b * Hc + h) * Sc + qt * 16 + l16) * HDc + half * 16;
    v16h aq[4];
#pragma unroll
    for (int kk = 0; kk < 4; ++kk) aq[kk] = *(const v16h*)(qbase + kk * 32);

    v8f out[8];
#pragma unroll
    for (int nt = 0; nt < 8; ++nt) out[nt] = (v8f){};
    float mrow[8], lrow[8];
#pragma unroll
    for (int r = 0; r < 8; ++r) { mrow[r] = -1e30f; lrow[r] = 0.0f; }

    const _Float16* kbase = Km + ((size_t)b * KVc + kvh) * Sc * HDc;
    const _Float16* vbase = Vt + ((size_t)b * KVc + kvh) * HDc * Sc;
    const int kmax = qt * 16 + 16;  // exclusive causal bound

    for (int kb = 0; kb * 32 < kmax; ++kb) {
        float p[2][8];
        // ---- scores: two 16x16 tiles over this 32-wide k block ----
#pragma unroll
        for (int t = 0; t < 2; ++t) {
            v8f s = {};
            const _Float16* kp = kbase + (size_t)(kb * 32 + t * 16 + l16) * HDc + half * 16;
#pragma unroll
            for (int kk = 0; kk < 4; ++kk) {
                v16h bk = *(const v16h*)(kp + kk * 32);
                s = wmma16(aq[kk], bk, s);
            }
            const int col = kb * 32 + t * 16 + l16;
#pragma unroll
            for (int r = 0; r < 8; ++r) {
                const int row = qt * 16 + r + 8 * half;
                float v = s[r] * RSQRT_HD;
                p[t][r] = (col <= row) ? v : -1e30f;
            }
        }
        // ---- online softmax (rows live in lane halves; width-16 reductions) ----
        float corr[8];
#pragma unroll
        for (int r = 0; r < 8; ++r) {
            float v = fmaxf(p[0][r], p[1][r]);
            v = fmaxf(v, __shfl_xor(v, 1, 16));
            v = fmaxf(v, __shfl_xor(v, 2, 16));
            v = fmaxf(v, __shfl_xor(v, 4, 16));
            v = fmaxf(v, __shfl_xor(v, 8, 16));
            const float mnew = fmaxf(mrow[r], v);
            corr[r] = __expf(mrow[r] - mnew);
            const float e0 = __expf(p[0][r] - mnew);
            const float e1 = __expf(p[1][r] - mnew);
            float sum = e0 + e1;
            sum += __shfl_xor(sum, 1, 16);
            sum += __shfl_xor(sum, 2, 16);
            sum += __shfl_xor(sum, 4, 16);
            sum += __shfl_xor(sum, 8, 16);
            lrow[r] = lrow[r] * corr[r] + sum;
            mrow[r] = mnew;
            p[0][r] = e0;
            p[1][r] = e1;
        }
#pragma unroll
        for (int nt = 0; nt < 8; ++nt)
#pragma unroll
            for (int r = 0; r < 8; ++r) out[nt][r] *= corr[r];

        // ---- reshape P (C-layout) -> A-layout via LDS ----
#pragma unroll
        for (int t = 0; t < 2; ++t)
#pragma unroll
            for (int r = 0; r < 8; ++r)
                lpw[(r + 8 * half) * 32 + t * 16 + l16] = (_Float16)p[t][r];
        v16h pa = *(const v16h*)(&lpw[l16 * 32 + half * 16]);

        // ---- PV: P(16x32) x V(32x128) ----
#pragma unroll
        for (int nt = 0; nt < 8; ++nt) {
            const _Float16* vp = vbase + (size_t)(nt * 16 + l16) * Sc + kb * 32 + half * 16;
            v16h bv = *(const v16h*)vp;
            out[nt] = wmma16(pa, bv, out[nt]);
        }
    }

    // ---- finalize: divide by l, write attn [token, H*HD] f16 ----
#pragma unroll
    for (int nt = 0; nt < 8; ++nt)
#pragma unroll
        for (int r = 0; r < 8; ++r) {
            const int row = qt * 16 + r + 8 * half;
            const float val = out[nt][r] / lrow[r];
            O[((size_t)b * Sc + row) * QNc + h * HDc + nt * 16 + l16] = (_Float16)val;
        }
}

// ---------------- SiLU(gate)*up + per-token fake-quant -> f16 ----------------
// one block (256 thr) per token row of length I
__global__ void k_silu_mul_quant(const float* __restrict__ g, const float* __restrict__ u,
                                 _Float16* __restrict__ out, int I) {
    __shared__ float smn[256], smx[256];
    const size_t base = (size_t)blockIdx.x * I;
    float mn = 3e38f, mx = -3e38f;
    for (int i = threadIdx.x; i < I; i += blockDim.x) {
        const float gv = g[base + i];
        const float v = gv * (1.0f / (1.0f + __expf(-gv))) * u[base + i];
        mn = fminf(mn, v); mx = fmaxf(mx, v);
    }
    blockMinMax(mn, mx, smn, smx);
    float scale, zp, inv;
    qparams(mn, mx, scale, zp, inv);
    for (int i = threadIdx.x; i < I; i += blockDim.x) {
        const float gv = g[base + i];
        const float v = gv * (1.0f / (1.0f + __expf(-gv))) * u[base + i];
        out[base + i] = (_Float16)qdq(v, scale, zp, inv);
    }
}

// ---------------- residual add ----------------
__global__ void k_add(const float* __restrict__ a, const float* __restrict__ b,
                      float* __restrict__ c, size_t n) {
    for (size_t i = blockIdx.x * (size_t)blockDim.x + threadIdx.x; i < n;
         i += (size_t)gridDim.x * blockDim.x)
        c[i] = a[i] + b[i];
}

// =========================================================================
extern "C" void kernel_launch(void* const* d_in, const int* in_sizes, int n_in,
                              void* d_out, int out_size, void* d_ws, size_t ws_size,
                              hipStream_t stream) {
    const float* hidden = (const float*)d_in[0];
    const float* q_w    = (const float*)d_in[1];
    const float* k_w    = (const float*)d_in[2];
    const float* v_w    = (const float*)d_in[3];
    const float* o_w    = (const float*)d_in[4];
    const float* gate_w = (const float*)d_in[5];
    const float* up_w   = (const float*)d_in[6];
    const float* down_w = (const float*)d_in[7];
    const float* ln1_w  = (const float*)d_in[8];
    const float* ln2_w  = (const float*)d_in[9];
    float* outp = (float*)d_out;

    char* base = (char*)d_ws;
    size_t off = 0;
    auto alloc = [&](size_t bytes) -> void* {
        void* r = base + off;
        off = (off + bytes + 255) & ~(size_t)255;
        return r;
    };

    _Float16* wq_q = (_Float16*)alloc((size_t)QNc * Dc * 2);
    _Float16* wq_k = (_Float16*)alloc((size_t)KNc * Dc * 2);
    _Float16* wq_v = (_Float16*)alloc((size_t)KNc * Dc * 2);
    _Float16* wq_o = (_Float16*)alloc((size_t)Dc * QNc * 2);
    _Float16* wq_g = (_Float16*)alloc((size_t)Ic * Dc * 2);
    _Float16* wq_u = (_Float16*)alloc((size_t)Ic * Dc * 2);
    _Float16* wq_d = (_Float16*)alloc((size_t)Dc * Ic * 2);
    _Float16* x1   = (_Float16*)alloc((size_t)Mc * Dc * 2);      // normed acts (reused pre/post attn)
    float*    qf32 = (float*)alloc((size_t)Mc * QNc * 4);
    float*    kf32 = (float*)alloc((size_t)Mc * KNc * 4);
    float*    vf32 = (float*)alloc((size_t)Mc * KNc * 4);
    _Float16* Qh   = (_Float16*)alloc((size_t)Mc * QNc * 2);
    _Float16* Kh   = (_Float16*)alloc((size_t)Bc * KVc * Sc * HDc * 2);
    _Float16* Vt   = (_Float16*)alloc((size_t)Bc * KVc * HDc * Sc * 2);
    _Float16* attn = (_Float16*)alloc((size_t)Mc * QNc * 2);
    float*    of32 = (float*)alloc((size_t)Mc * Dc * 4);
    float*    h1   = (float*)alloc((size_t)Mc * Dc * 4);
    float*    g32  = (float*)alloc((size_t)Mc * Ic * 4);
    float*    u32  = (float*)alloc((size_t)Mc * Ic * 4);
    _Float16* mh   = (_Float16*)alloc((size_t)Mc * Ic * 2);
    float*    dn32 = (float*)alloc((size_t)Mc * Dc * 4);
    (void)ws_size; (void)in_sizes; (void)n_in; (void)out_size;

    // 1) quantize all weights once (per output channel) -> f16
    k_quant_weight<<<QNc, 256, 0, stream>>>(q_w,    wq_q, Dc);
    k_quant_weight<<<KNc, 256, 0, stream>>>(k_w,    wq_k, Dc);
    k_quant_weight<<<KNc, 256, 0, stream>>>(v_w,    wq_v, Dc);
    k_quant_weight<<<Dc,  256, 0, stream>>>(o_w,    wq_o, QNc);
    k_quant_weight<<<Ic,  256, 0, stream>>>(gate_w, wq_g, Dc);
    k_quant_weight<<<Ic,  256, 0, stream>>>(up_w,   wq_u, Dc);
    k_quant_weight<<<Dc,  256, 0, stream>>>(down_w, wq_d, Ic);

    // 2) RMSNorm + per-token quant
    k_rmsnorm_quant<<<Mc, 256, 0, stream>>>(hidden, ln1_w, x1, Dc);

    // 3) QKV projections (WMMA, async-LDS staged)
    k_gemm_tn<<<dim3(QNc / 128, Mc / 64), 128, 0, stream>>>(x1, wq_q, qf32, QNc, Dc);
    k_gemm_tn<<<dim3(KNc / 128, Mc / 64), 128, 0, stream>>>(x1, wq_k, kf32, KNc, Dc);
    k_gemm_tn<<<dim3(KNc / 128, Mc / 64), 128, 0, stream>>>(x1, wq_v, vf32, KNc, Dc);

    // 4) RoPE + quant -> f16 head-major; V quant + transpose
    k_rope_quant<<<dim3(Sc, Hc + KVc, Bc), 128, 0, stream>>>(qf32, kf32, Qh, Kh);
    k_vq_transpose<<<dim3(Sc, KVc, Bc), 128, 0, stream>>>(vf32, Vt);

    // 5) causal GQA attention (WMMA flash)
    k_attn<<<dim3(Sc / 64, Hc, Bc), 128, 0, stream>>>(Qh, Kh, Vt, attn);

    // 6) output projection + residual
    k_gemm_tn<<<dim3(Dc / 128, Mc / 64), 128, 0, stream>>>(attn, wq_o, of32, Dc, QNc);
    k_add<<<4096, 256, 0, stream>>>(hidden, of32, h1, (size_t)Mc * Dc);

    // 7) MLP: norm -> gate/up -> silu*up -> down -> residual
    k_rmsnorm_quant<<<Mc, 256, 0, stream>>>(h1, ln2_w, x1, Dc);
    k_gemm_tn<<<dim3(Ic / 128, Mc / 64), 128, 0, stream>>>(x1, wq_g, g32, Ic, Dc);
    k_gemm_tn<<<dim3(Ic / 128, Mc / 64), 128, 0, stream>>>(x1, wq_u, u32, Ic, Dc);
    k_silu_mul_quant<<<Mc, 256, 0, stream>>>(g32, u32, mh, Ic);
    k_gemm_tn<<<dim3(Dc / 128, Mc / 64), 128, 0, stream>>>(mh, wq_d, dn32, Dc, Ic);
    k_add<<<4096, 256, 0, stream>>>(h1, dn32, outp, (size_t)Mc * Dc);
}
